// self_learning_poisoner_24764781429353
// MI455X (gfx1250) — matrix operand order repaired and checked by reading, not compiled, not run on hardware
//
#include <hip/hip_runtime.h>
#include <hip/hip_bf16.h>
#include <stdint.h>

typedef __attribute__((ext_vector_type(16))) _Float16 v16h;
typedef __attribute__((ext_vector_type(8)))  float    v8f;

#define B_  64
#define L_  128
#define C_  5
#define E_  768
#define TPB 192   // 6 waves; one float4 (16B) chunk of the 768-elem row per thread
#define NW  6

__global__ __launch_bounds__(TPB)
void poisoner_fused(const int*   __restrict__ to_ids,
                    const int*   __restrict__ no_ids,
                    const int*   __restrict__ cand_ids,
                    const float* __restrict__ word_emb,
                    const float* __restrict__ pos_emb,
                    const float* __restrict__ rel_mat,
                    const float* __restrict__ rel_bias,
                    const float* __restrict__ gumbel,
                    float*       __restrict__ out)
{
    __shared__ float4   sCand[C_][TPB];      // candidate word rows   (15 KB)
    __shared__ float4   sTp[TPB];            // to_poison word row    (3 KB)
    __shared__ float4   sRel[TPB];           // relevance_mat[l]      (3 KB)
    __shared__ float4   sNo[TPB];            // no_poison word row    (3 KB)
    __shared__ float4   sPos[TPB];           // pos_emb[l]            (3 KB)
    __shared__ _Float16 sDiffH[C_ + 1][E_];  // f16 diffs; row C_ is all-zero pad
    __shared__ _Float16 sRelH[E_];           // f16 relevance vector for WMMA B
    __shared__ float    sPartial[NW][8];     // per-wave partial scores
    __shared__ float    sProb[C_];

    const int t    = threadIdx.x;
    const int lane = t & 31;
    const int wv   = t >> 5;
    const int bl   = blockIdx.x;          // b*L + l
    const int l    = bl & (L_ - 1);

    // ---------------- token ids for this (b,l) ----------------
    const int tp_id = to_ids[bl];
    const int no_id = no_ids[bl];
    int cid[C_];
    #pragma unroll
    for (int c = 0; c < C_; ++c) cid[c] = cand_ids[bl * C_ + c];

    // ---------------- Phase 0: async DMA of 9 rows into LDS ----------------
    // Each lane copies its 16B chunk of every row via global->LDS async copies
    // (ASYNCcnt-tracked, no VGPR staging).
    {
        const uint64_t gw    = (uint64_t)(uintptr_t)word_emb;
        const uint64_t chunk = (uint64_t)t * 16u;  // byte offset within a row
        uint64_t ga;
        uint32_t la;
        #pragma unroll
        for (int c = 0; c < C_; ++c) {
            ga = gw + (uint64_t)cid[c] * (E_ * 4) + chunk;
            la = (uint32_t)(uintptr_t)&sCand[c][t];
            asm volatile("global_load_async_to_lds_b128 %0, %1, off"
                         :: "v"(la), "v"(ga) : "memory");
        }
        ga = gw + (uint64_t)tp_id * (E_ * 4) + chunk;
        la = (uint32_t)(uintptr_t)&sTp[t];
        asm volatile("global_load_async_to_lds_b128 %0, %1, off" :: "v"(la), "v"(ga) : "memory");
        ga = gw + (uint64_t)no_id * (E_ * 4) + chunk;
        la = (uint32_t)(uintptr_t)&sNo[t];
        asm volatile("global_load_async_to_lds_b128 %0, %1, off" :: "v"(la), "v"(ga) : "memory");
        ga = (uint64_t)(uintptr_t)rel_mat + (uint64_t)l * (E_ * 4) + chunk;
        la = (uint32_t)(uintptr_t)&sRel[t];
        asm volatile("global_load_async_to_lds_b128 %0, %1, off" :: "v"(la), "v"(ga) : "memory");
        ga = (uint64_t)(uintptr_t)pos_emb + (uint64_t)l * (E_ * 4) + chunk;
        la = (uint32_t)(uintptr_t)&sPos[t];
        asm volatile("global_load_async_to_lds_b128 %0, %1, off" :: "v"(la), "v"(ga) : "memory");
        asm volatile("s_wait_asynccnt 0x0" ::: "memory");
    }
    __syncthreads();

    // ---------------- Phase 1a: build f16 A/B operands in LDS ----------------
    // Row C_ of sDiffH is zero-filled so WMMA A-rows 5..15 read constants from
    // LDS instead of per-element v_cndmask selects in the inner loop.
    {
        const float* candf = (const float*)&sCand[0][0];
        const float* tpf   = (const float*)&sTp[0];
        const float* relf  = (const float*)&sRel[0];
        #pragma unroll
        for (int r = 0; r < E_ / TPB; ++r) {
            const int e    = t + r * TPB;
            const float tv = tpf[e];
            sRelH[e] = (_Float16)relf[e];
            #pragma unroll
            for (int c = 0; c < C_; ++c)
                sDiffH[c][e] = (_Float16)(candf[c * E_ + e] - tv);
            sDiffH[C_][e] = (_Float16)0.0f;   // zero pad row
        }
    }
    __syncthreads();

    // ---------------- Phase 1b: scores via v_wmma_f32_16x16x32_f16 ----------------
    // A rows 0..4 = diff vectors (rows 5..15 read the zero row); B replicates
    // the relevance vector across all 16 columns. K=768 -> 24 steps, 4/wave.
    {
        const int m    = lane & 15;
        const int hi   = lane >> 4;
        const int msel = (m < C_) ? m : C_;   // invalid rows -> zero pad row
        v8f acc = {};
        #pragma unroll
        for (int s = 0; s < 4; ++s) {
            const int kbase = (wv + s * NW) * 32;
            v16h a, bm;
            #pragma unroll
            for (int j = 0; j < 8; ++j) {
                // 16-bit A 16x32 layout: VGPR j holds K = (j<4 ? 2j : 8+2j) (+8 hi half)
                const int ka = kbase + ((j < 4) ? 2 * j : 8 + 2 * j) + (hi ? 8 : 0);
                a[2 * j]     = sDiffH[msel][ka];
                a[2 * j + 1] = sDiffH[msel][ka + 1];
                // B 32x16: same K order as A (same value all cols)
                const int kb = kbase + ((j < 4) ? 2 * j : 8 + 2 * j) + (hi ? 8 : 0);
                bm[2 * j]     = sRelH[kb];
                bm[2 * j + 1] = sRelH[kb + 1];
            }
            acc = __builtin_amdgcn_wmma_f32_16x16x32_f16(
                false, a, false, bm, (short)0, acc, false, false);
        }
        // D[m][n]: lane n (0-15) holds rows 0-7 in vector elems 0-7 -> lane 0 has scores 0..4
        if (lane == 0) {
            #pragma unroll
            for (int c = 0; c < C_; ++c) sPartial[wv][c] = acc[c];
        }
    }
    __syncthreads();

    // ---------------- Phase 1c: gumbel-softmax over C=5 ----------------
    if (t == 0) {
        float y[C_];
        float mx = -1e30f;
        #pragma unroll
        for (int c = 0; c < C_; ++c) {
            float s = 0.f;
            #pragma unroll
            for (int w = 0; w < NW; ++w) s += sPartial[w][c];
            const float logit = s + rel_bias[l * C_ + c] + gumbel[bl * C_ + c];
            y[c] = logit * 2.0f;             // (logits + g) / 0.5
            mx = fmaxf(mx, y[c]);
        }
        float denom = 0.f;
        #pragma unroll
        for (int c = 0; c < C_; ++c) { y[c] = __expf(y[c] - mx); denom += y[c]; }
        const float inv = 1.0f / denom;
        #pragma unroll
        for (int c = 0; c < C_; ++c) sProb[c] = y[c] * inv;
    }
    __syncthreads();

    // ---------------- Phase 2: blend + both output streams ----------------
    {
        float p[C_];
        #pragma unroll
        for (int c = 0; c < C_; ++c) p[c] = sProb[c];
        const float4 pos4 = sPos[t];
        float4 acc4 = make_float4(0.f, 0.f, 0.f, 0.f);
        #pragma unroll
        for (int c = 0; c < C_; ++c) {
            const float4 w4 = sCand[c][t];
            acc4.x += p[c] * (w4.x + pos4.x);
            acc4.y += p[c] * (w4.y + pos4.y);
            acc4.z += p[c] * (w4.z + pos4.z);
            acc4.w += p[c] * (w4.w + pos4.w);
        }
        float4* outv = (float4*)out;
        outv[bl * (E_ / 4) + t] = acc4;                       // poisoned stream [b,l]
        const float4 n4 = sNo[t];
        float4 clean;
        clean.x = n4.x + pos4.x; clean.y = n4.y + pos4.y;
        clean.z = n4.z + pos4.z; clean.w = n4.w + pos4.w;
        outv[(B_ * L_ + bl) * (E_ / 4) + t] = clean;          // clean stream [B+b,l]
    }
}

extern "C" void kernel_launch(void* const* d_in, const int* in_sizes, int n_in,
                              void* d_out, int out_size, void* d_ws, size_t ws_size,
                              hipStream_t stream) {
    const int*   to_ids   = (const int*)  d_in[0];
    const int*   no_ids   = (const int*)  d_in[1];
    const int*   cand_ids = (const int*)  d_in[2];
    const float* word_emb = (const float*)d_in[3];
    const float* pos_emb  = (const float*)d_in[4];
    const float* rel_mat  = (const float*)d_in[5];
    const float* rel_bias = (const float*)d_in[6];
    const float* gumbel   = (const float*)d_in[7];
    float*       out      = (float*)d_out;

    dim3 grid(B_ * L_), block(TPB);
    poisoner_fused<<<grid, block, 0, stream>>>(to_ids, no_ids, cand_ids, word_emb,
                                               pos_emb, rel_mat, rel_bias, gumbel, out);
}
